// TransformerBlock_12841952215351
// MI455X (gfx1250) — matrix operand order repaired and checked by reading, not compiled, not run on hardware
//
#include <hip/hip_runtime.h>
#include <hip/hip_bf16.h>

// ---------------------------------------------------------------------------
// TransformerBlock on MI455X (gfx1250).
// All GEMMs: v_wmma_f32_16x16x32_bf16, f32 accumulate.
// All WMMA operands stored fragment-packed bf16: per tile (16x32 A / 32x16 B),
// lane L owns 16 contiguous bf16 (32 B) -> two b128 loads per fragment.
// DIM=128, N=32, S=1024, M = 32768 rows.
// ---------------------------------------------------------------------------

#define DIM   128
#define SEQ   1024
#define BATCH 32
#define MROWS (BATCH * SEQ)

typedef __bf16 bf16_t;
typedef __attribute__((ext_vector_type(16))) __bf16 v16bf;
typedef __attribute__((ext_vector_type(8)))  float  v8f;

static __device__ inline bf16_t to_bf(float f) { return (bf16_t)f; }

static __device__ inline v8f wmma_bf16(v16bf a, v16bf b, v8f c) {
  return __builtin_amdgcn_wmma_f32_16x16x32_bf16(false, a, false, b,
                                                 (short)0, c, false, false);
}

// Packed-fragment load: tile = 32 lanes x 16 bf16 (1024 B), lane-contiguous.
static __device__ inline v16bf load_frag_pk(const bf16_t* __restrict__ base, size_t tile) {
  return *(const v16bf*)(base + (tile * 32 + (threadIdx.x & 31)) * 16);
}

// A-matrix 16x32 fragment from f32 (row-major, ld floats) -- used for LDS data.
static __device__ inline v16bf load_a16x32_f32(const float* src, int ld) {
  const int lane = threadIdx.x & 31;
  const int m    = lane & 15;
  const int kb   = (lane >> 4) << 3;
  const float* row = src + (size_t)m * ld;
  v16bf a;
#pragma unroll
  for (int i = 0; i < 8; ++i) a[i]     = to_bf(row[kb + i]);
#pragma unroll
  for (int i = 0; i < 8; ++i) a[8 + i] = to_bf(row[16 + kb + i]);
  return a;
}

// ---- packed-layout scatter helpers (element -> lane/elem maps) ----
// A-pack element (m,k) of 16x32 tile:
//   hi=((k&15)>=8); lane=m+16*hi; elem=(k<16)? k-8*hi : k-8-8*hi
static __device__ inline void store_apack(bf16_t* base, int rowTile, int kstep,
                                          int m, int k, bf16_t h) {
  const int hi = ((k & 15) >= 8) ? 1 : 0;
  const int lp = m + 16 * hi;
  const int e  = (k < 16) ? (k - 8 * hi) : (k - 8 - 8 * hi);
  base[(((size_t)rowTile * 4 + kstep) * 32 + lp) * 16 + e] = h;
}
// Kt-pack (B = K-tile^T): lane = m + 16*((k>>3)&1); elem = (k&7) + 8*(k>>4)
static __device__ inline void store_ktpack(bf16_t* base, int rowTile, int kstep,
                                           int m, int k, bf16_t h) {
  const int lp = m + 16 * ((k >> 3) & 1);
  base[(((size_t)rowTile * 4 + kstep) * 32 + lp) * 16 + ((k & 7) + ((k >> 4) << 3))] = h;
}
// B-pack element (k of 32, n of 16): lane = n + 16*((k>>3)&1); elem = (k&7) + 8*(k>>4)
static __device__ inline void store_bpack(bf16_t* base, size_t tile,
                                          int k, int n, bf16_t h) {
  const int lp = n + 16 * ((k >> 3) & 1);
  base[(tile * 32 + lp) * 16 + ((k & 7) + ((k >> 4) << 3))] = h;
}

// ---------------------------------------------------------------------------
// Pack kernels (run once; weights reused 2048x, x read 3x)
// ---------------------------------------------------------------------------
__global__ __launch_bounds__(256) void pack_a_kernel(const float* __restrict__ src,
                                                     bf16_t* __restrict__ dst,
                                                     int Mtiles, int Ktiles, int ld) {
  const int tile = blockIdx.x * 8 + (threadIdx.x >> 5);
  if (tile >= Mtiles * Ktiles) return;
  const int tm = tile / Ktiles, tk = tile % Ktiles;
  const int lane = threadIdx.x & 31;
  const int m  = lane & 15;
  const int kb = (lane >> 4) << 3;
  const float* row = src + (size_t)(tm * 16 + m) * ld + tk * 32;
  v16bf a;
#pragma unroll
  for (int i = 0; i < 8; ++i) a[i]     = to_bf(row[kb + i]);
#pragma unroll
  for (int i = 0; i < 8; ++i) a[8 + i] = to_bf(row[16 + kb + i]);
  *(v16bf*)(dst + ((size_t)tile * 32 + lane) * 16) = a;
}

__global__ __launch_bounds__(256) void pack_b_kernel(const float* __restrict__ src,
                                                     bf16_t* __restrict__ dst,
                                                     int Ktiles, int Ntiles, int ld) {
  const int tile = blockIdx.x * 8 + (threadIdx.x >> 5);
  if (tile >= Ktiles * Ntiles) return;
  const int tk = tile / Ntiles, tn = tile % Ntiles;
  const int lane = threadIdx.x & 31;
  const int n  = lane & 15;
  const int kb = (lane >> 4) << 3;
  const float* base = src + (size_t)(tk * 32 + kb) * ld + tn * 16 + n;
  v16bf b;
#pragma unroll
  for (int e = 0; e < 16; ++e) b[e] = to_bf(base[(size_t)(e + ((e >> 3) << 3)) * ld]);
  *(v16bf*)(dst + ((size_t)tile * 32 + lane) * 16) = b;
}

// ---------------------------------------------------------------------------
// Kernel 1: qkv = x @ w_qkv + b_qkv, outputs routed straight into the
// consumer-packed layouts: q->A-pack, k->Kt-pack, v->B-pack (all bf16).
// ---------------------------------------------------------------------------
__global__ __launch_bounds__(256) void qkv_kernel(const bf16_t* __restrict__ xpk,
                                                  const bf16_t* __restrict__ wpk,
                                                  const float* __restrict__ bias,
                                                  bf16_t* __restrict__ qpk,
                                                  bf16_t* __restrict__ kpk,
                                                  bf16_t* __restrict__ vpk) {
  const int blk    = blockIdx.x;
  const int rt     = blk / 3;
  const int colGrp = blk % 3;
  const int wave   = threadIdx.x >> 5;
  const int lane   = threadIdx.x & 31;
  const int nt     = colGrp * 8 + wave; // 0..23
  const int n0     = nt * 16;

  v8f c = {};
#pragma unroll
  for (int kk = 0; kk < 4; ++kk) {
    v16bf af = load_frag_pk(xpk, rt * 4 + kk);
    v16bf bf = load_frag_pk(wpk, kk * 24 + nt);
    c = wmma_bf16(af, bf, c);
  }
  const int n  = lane & 15;
  const int mb = (lane >> 4) * 8;
  const int ng = n0 + n;                // whole wave in one of q/k/v ranges
  const float bv = bias[ng];
#pragma unroll
  for (int i = 0; i < 8; ++i) {
    const bf16_t h = to_bf(c[i] + bv);
    const int rg = rt * 16 + mb + i;
    if (ng < 128) {
      store_apack(qpk, rg >> 4, ng >> 5, rg & 15, ng & 31, h);
    } else if (ng < 256) {
      const int f = ng - 128;
      store_ktpack(kpk, rg >> 4, f >> 5, rg & 15, f & 31, h);
    } else {
      const int d = ng - 256;
      const int s = rg & (SEQ - 1), batch = rg >> 10;
      const size_t tile = ((size_t)batch * 32 + (s >> 5)) * 8 + (d >> 4);
      store_bpack(vpk, tile, s & 31, d & 15, h);
    }
  }
}

// ---------------------------------------------------------------------------
// Kernel 2: attention.  One block per (batch, 16-query tile).
// 16x1024 f32 score strip in LDS (64 KB); after the register-resident softmax
// the SAME LDS is reused as A-frag-packed bf16 probabilities, so P@V loads
// each fragment as one contiguous 32 B per-lane read (no gathers, no cvt).
// ---------------------------------------------------------------------------
__global__ __launch_bounds__(256) void attn_kernel(const bf16_t* __restrict__ qpk,
                                                   const bf16_t* __restrict__ kpk,
                                                   const bf16_t* __restrict__ vpk,
                                                   bf16_t* __restrict__ apk) {
  const int blk   = blockIdx.x;        // global 16-row tile index rt
  const int batch = blk >> 6;
  const int tid   = threadIdx.x;
  const int wave  = tid >> 5;
  const int lane  = tid & 31;

  __shared__ __align__(32) float sc[16][SEQ];          // 64 KB
  bf16_t* pb = reinterpret_cast<bf16_t*>(&sc[0][0]);   // aliased after softmax

  // Q fragments, reused for all 64 key tiles.
  v16bf qa[4];
#pragma unroll
  for (int kk = 0; kk < 4; ++kk) qa[kk] = load_frag_pk(qpk, blk * 4 + kk);

  const float scale = 0.08838834764831845f;   // 1/sqrt(128)

  // ---- scores: scale * Q @ K^T -> LDS ----
  for (int j = wave; j < SEQ / 16; j += 8) {
    if (j + 8 < SEQ / 16)
      __builtin_prefetch(kpk + ((size_t)((batch * 64 + j + 8) * 4) * 32) * 16, 0, 1);
    v8f c = {};
#pragma unroll
    for (int kk = 0; kk < 4; ++kk) {
      v16bf bf = load_frag_pk(kpk, (batch * 64 + j) * 4 + kk);
      c = wmma_bf16(qa[kk], bf, c);
    }
    const int n  = lane & 15;
    const int mb = (lane >> 4) * 8;
#pragma unroll
    for (int i = 0; i < 8; ++i) sc[mb + i][j * 16 + n] = c[i] * scale;
  }
  __syncthreads();

  // ---- softmax: scores pulled once into registers; shuffle reductions;
  //      normalized probs written back A-frag-packed bf16 (aliasing sc) ----
  {
    const int row = tid >> 4;
    const int sub = tid & 15;
    float e[SEQ / 16];
    float mx = -3.0e38f;
#pragma unroll
    for (int j = 0; j < SEQ / 16; ++j) {
      e[j] = sc[row][sub + 16 * j];
      mx = fmaxf(mx, e[j]);
    }
#pragma unroll
    for (int off = 1; off < 16; off <<= 1) mx = fmaxf(mx, __shfl_xor(mx, off, 32));
    float sum = 0.f;
#pragma unroll
    for (int j = 0; j < SEQ / 16; ++j) {
      e[j] = __expf(e[j] - mx);
      sum += e[j];
    }
#pragma unroll
    for (int off = 1; off < 16; off <<= 1) sum += __shfl_xor(sum, off, 32);
    const float inv = 1.0f / sum;
    __syncthreads();   // everyone done reading raw scores before overwrite
#pragma unroll
    for (int j = 0; j < SEQ / 16; ++j) {
      const int cg = sub + 16 * j;          // global key column
      const int t  = cg >> 5, k = cg & 31;  // A-frag tile / k within tile
      const int hi = ((k & 15) >= 8) ? 1 : 0;
      const int lp = row + 16 * hi;
      const int ee = (k < 16) ? (k - 8 * hi) : (k - 8 - 8 * hi);
      pb[((size_t)t * 32 + lp) * 16 + ee] = to_bf(e[j] * inv);
    }
  }
  __syncthreads();

  // ---- O = P @ V : wave w owns output cols [16w,16w+16) ----
  v8f o = {};
  for (int ks = 0; ks < SEQ; ks += 32) {
    __builtin_prefetch(
        vpk + (((size_t)batch * 32 + ((ks >> 5) + 1)) * 8 + wave) * 32 * 16, 0, 1);
    v16bf pa = *(const v16bf*)(pb + ((size_t)(ks >> 5) * 32 + lane) * 16);
    v16bf vb = load_frag_pk(vpk, ((size_t)batch * 32 + (ks >> 5)) * 8 + wave);
    o = wmma_bf16(pa, vb, o);
  }
  // write A-packed for the proj GEMM
  const int n  = lane & 15;
  const int mb = (lane >> 4) * 8;
  const int d0 = wave * 16;
#pragma unroll
  for (int i = 0; i < 8; ++i)
    store_apack(apk, blk, (d0 + n) >> 5, mb + i, (d0 + n) & 31, to_bf(o[i]));
}

// ---------------------------------------------------------------------------
// Kernel 3: h = attn @ w_proj + b_proj ; out = LayerNorm(x + h)
// ---------------------------------------------------------------------------
__global__ __launch_bounds__(256) void proj_ln1_kernel(const bf16_t* __restrict__ apk,
                                                       const float* __restrict__ x,
                                                       const bf16_t* __restrict__ wpk,
                                                       const float* __restrict__ b_proj,
                                                       const float* __restrict__ g,
                                                       const float* __restrict__ bvec,
                                                       float* __restrict__ out) {
  const int rt   = blockIdx.x;
  const int m0   = rt * 16;
  const int wave = threadIdx.x >> 5;
  const int lane = threadIdx.x & 31;
  const int n0   = wave * 16;

  __shared__ float h[16][DIM];

  v8f c = {};
#pragma unroll
  for (int kk = 0; kk < 4; ++kk) {
    v16bf af = load_frag_pk(apk, rt * 4 + kk);
    v16bf bf = load_frag_pk(wpk, kk * 8 + wave);
    c = wmma_bf16(af, bf, c);
  }
  {
    const int n  = lane & 15;
    const int mb = (lane >> 4) * 8;
    const float bv = b_proj[n0 + n];
#pragma unroll
    for (int i = 0; i < 8; ++i) h[mb + i][n0 + n] = c[i] + bv;
  }
  __syncthreads();

  const int row = threadIdx.x >> 4;
  const int sub = threadIdx.x & 15;
  const float* xr = x + (size_t)(m0 + row) * DIM;
  float vals[8];
  float s = 0.f;
#pragma unroll
  for (int i = 0; i < 8; ++i) {
    const int col = sub + 16 * i;
    const float v = xr[col] + h[row][col];
    vals[i] = v;
    s += v;
  }
#pragma unroll
  for (int off = 1; off < 16; off <<= 1) s += __shfl_xor(s, off, 32);
  const float mu = s * (1.0f / DIM);
  float vs = 0.f;
#pragma unroll
  for (int i = 0; i < 8; ++i) { const float d = vals[i] - mu; vs += d * d; }
#pragma unroll
  for (int off = 1; off < 16; off <<= 1) vs += __shfl_xor(vs, off, 32);
  const float rstd = rsqrtf(vs * (1.0f / DIM) + 1e-5f);
  float* dst = out + (size_t)(m0 + row) * DIM;
#pragma unroll
  for (int i = 0; i < 8; ++i) {
    const int col = sub + 16 * i;
    dst[col] = (vals[i] - mu) * rstd * g[col] + bvec[col];
  }
}

// ---------------------------------------------------------------------------
// Kernel 4: 6-layer MLP (ReLU) + residual + LN2; LDS ping-pong per 16-row tile.
// ---------------------------------------------------------------------------
__global__ __launch_bounds__(256) void mlp_ln2_kernel(const float* __restrict__ x1,
                                                      const bf16_t* __restrict__ wpk,
                                                      const float* __restrict__ ff_b,
                                                      const float* __restrict__ g,
                                                      const float* __restrict__ bvec,
                                                      float* __restrict__ out) {
  const int m0   = blockIdx.x * 16;
  const int wave = threadIdx.x >> 5;
  const int lane = threadIdx.x & 31;
  const int n0   = wave * 16;

  __shared__ float buf0[16][DIM];
  __shared__ float buf1[16][DIM];

  {
    const int row = threadIdx.x >> 4;
    const int sub = threadIdx.x & 15;
    const float* xr = x1 + (size_t)(m0 + row) * DIM;
#pragma unroll
    for (int i = 0; i < 8; ++i) buf0[row][sub + 16 * i] = xr[sub + 16 * i];
  }
  __syncthreads();

  float (*bin)[DIM]  = buf0;
  float (*bout)[DIM] = buf1;

  for (int layer = 0; layer < 6; ++layer) {
    v8f c = {};
#pragma unroll
    for (int kk = 0; kk < 4; ++kk) {
      v16bf af = load_a16x32_f32(&bin[0][0] + kk * 32, DIM);
      v16bf bf = load_frag_pk(wpk, (layer * 4 + kk) * 8 + wave);
      c = wmma_bf16(af, bf, c);
    }
    const int n  = lane & 15;
    const int mb = (lane >> 4) * 8;
    const float bv = ff_b[layer * DIM + n0 + n];
    const bool relu = layer < 5;
#pragma unroll
    for (int i = 0; i < 8; ++i) {
      const float v = c[i] + bv;
      bout[mb + i][n0 + n] = relu ? fmaxf(v, 0.f) : v;
    }
    __syncthreads();
    float (*t)[DIM] = bin; bin = bout; bout = t;
  }

  const int row = threadIdx.x >> 4;
  const int sub = threadIdx.x & 15;
  const float* xr = x1 + (size_t)(m0 + row) * DIM;
  float vals[8];
  float s = 0.f;
#pragma unroll
  for (int i = 0; i < 8; ++i) {
    const int col = sub + 16 * i;
    const float v = xr[col] + bin[row][col];
    vals[i] = v;
    s += v;
  }
#pragma unroll
  for (int off = 1; off < 16; off <<= 1) s += __shfl_xor(s, off, 32);
  const float mu = s * (1.0f / DIM);
  float vs = 0.f;
#pragma unroll
  for (int i = 0; i < 8; ++i) { const float d = vals[i] - mu; vs += d * d; }
#pragma unroll
  for (int off = 1; off < 16; off <<= 1) vs += __shfl_xor(vs, off, 32);
  const float rstd = rsqrtf(vs * (1.0f / DIM) + 1e-5f);
  float* dst = out + (size_t)(m0 + row) * DIM;
#pragma unroll
  for (int i = 0; i < 8; ++i) {
    const int col = sub + 16 * i;
    dst[col] = (vals[i] - mu) * rstd * g[col] + bvec[col];
  }
}

// ---------------------------------------------------------------------------
extern "C" void kernel_launch(void* const* d_in, const int* in_sizes, int n_in,
                              void* d_out, int out_size, void* d_ws, size_t ws_size,
                              hipStream_t stream) {
  const float* x      = (const float*)d_in[0];
  const float* w_qkv  = (const float*)d_in[1];
  const float* b_qkv  = (const float*)d_in[2];
  const float* w_proj = (const float*)d_in[3];
  const float* b_proj = (const float*)d_in[4];
  const float* ln1_g  = (const float*)d_in[5];
  const float* ln1_b  = (const float*)d_in[6];
  const float* ln2_g  = (const float*)d_in[7];
  const float* ln2_b  = (const float*)d_in[8];
  const float* ff_w   = (const float*)d_in[9];
  const float* ff_b   = (const float*)d_in[10];
  float* out = (float*)d_out;

  unsigned char* ws = (unsigned char*)d_ws;
  const size_t SZ_F32 = (size_t)MROWS * DIM * sizeof(float);
  const size_t SZ_PK  = (size_t)MROWS * DIM * sizeof(bf16_t);
  float*  x1   = (float*)ws;                       ws += SZ_F32;
  bf16_t* xpk  = (bf16_t*)ws;                      ws += SZ_PK;
  bf16_t* qpk  = (bf16_t*)ws;                      ws += SZ_PK;
  bf16_t* kpk  = (bf16_t*)ws;                      ws += SZ_PK;
  bf16_t* vpk  = (bf16_t*)ws;                      ws += SZ_PK;
  bf16_t* apk  = (bf16_t*)ws;                      ws += SZ_PK;
  bf16_t* wqkv_pk  = (bf16_t*)ws;                  ws += (size_t)96  * 512 * 2;
  bf16_t* wproj_pk = (bf16_t*)ws;                  ws += (size_t)32  * 512 * 2;
  bf16_t* ffw_pk   = (bf16_t*)ws;

  // --- pack passes (cheap, run once) ---
  pack_a_kernel<<<(MROWS / 16) * 4 / 8, 256, 0, stream>>>(x, xpk, MROWS / 16, 4, DIM);
  pack_b_kernel<<<(96 + 7) / 8,  256, 0, stream>>>(w_qkv,  wqkv_pk,  4, 24, 3 * DIM);
  pack_b_kernel<<<(32 + 7) / 8,  256, 0, stream>>>(w_proj, wproj_pk, 4, 8,  DIM);
  pack_b_kernel<<<(192 + 7) / 8, 256, 0, stream>>>(ff_w,   ffw_pk,   24, 8, DIM);

  // --- transformer block ---
  qkv_kernel     <<<(MROWS / 16) * 3, 256, 0, stream>>>(xpk, wqkv_pk, b_qkv,
                                                        qpk, kpk, vpk);
  attn_kernel    <<<BATCH * (SEQ / 16), 256, 0, stream>>>(qpk, kpk, vpk, apk);
  proj_ln1_kernel<<<MROWS / 16, 256, 0, stream>>>(apk, x, wproj_pk, b_proj,
                                                  ln1_g, ln1_b, x1);
  mlp_ln2_kernel <<<MROWS / 16, 256, 0, stream>>>(x1, ffw_pk, ff_b,
                                                  ln2_g, ln2_b, out);
}
